// Transformer_71897752535206
// MI455X (gfx1250) — matrix operand order repaired and checked by reading, not compiled, not run on hardware
//
#include <hip/hip_runtime.h>
#include <hip/hip_bf16.h>

// ---------------- dims (match reference) ----------------
#define N_RAW   20000
#define BS      64
#define IN_DIM  16
#define CD      3
#define HDIM    32
#define H       8
#define NL      4
#define NH      3
#define KW      8
#define NPAD    20032          // ceil(20000/64)*64 ; every GEMM M == NPAD (multiple of 64)
#define NBLK    313            // NPAD/64
#define HD2     16
#define MH      256
#define SORTN   32768          // next pow2 >= NPAD
#define NSLICE  48             // 24 q-slices + 24 k-slices

typedef __attribute__((ext_vector_type(16))) _Float16 v16h;
typedef __attribute__((ext_vector_type(8)))  _Float16 v8h;
typedef __attribute__((ext_vector_type(4)))  _Float16 v4h;
typedef __attribute__((ext_vector_type(8)))  float    v8f;
typedef __attribute__((ext_vector_type(4)))  int      v4i;

__device__ __forceinline__ v16h cat16(v8h lo, v8h hi) {
  return __builtin_shufflevector(lo, hi, 0, 1, 2, 3, 4, 5, 6, 7, 8, 9, 10, 11, 12, 13, 14, 15);
}

#if defined(__has_builtin)
#  if __has_builtin(__builtin_amdgcn_global_load_async_to_lds_b128)
#    define HAVE_ASYNC_LDS 1
#  endif
#endif
#ifndef HAVE_ASYNC_LDS
#  define HAVE_ASYNC_LDS 0
#endif

#if HAVE_ASYNC_LDS
#define ASYNC_B128(gsrc, ldst)                                                  \
  __builtin_amdgcn_global_load_async_to_lds_b128(                               \
      (__attribute__((address_space(1))) v4i*)(gsrc),                           \
      (__attribute__((address_space(3))) v4i*)(ldst), 0, 0)
#endif

// ---------------- input index map (setup_inputs dict order; params pytree sorted-key order) ----
// 0:x 1:coords 2:edge_index(i64,unused) 3:batch(i64,unused) 4:bin_e(i32) 5:bin_p(i32)
// params: 6:W 7:e1 8:e1b 9:e2 10:e2b
// 11+15*L + {0:alpha 1:beta 2:f1 3:f1b 4:f2 5:f2b 6:n1b 7:n1g 8:n2b 9:n2g 10:w_rpe 11:wk 12:wo 13:wq 14:wv}
// 71..75:mlp_b[0..4] 76..79:mlp_bt[0..3] 80..83:mlp_g[0..3] 84..88:mlp_w[0..4]
#define IX_W 6
#define IX_E1 7
#define IX_E1B 8
#define IX_E2 9
#define IX_E2B 10
#define IX_LAYER(L) (11 + 15*(L))
#define LO_ALPHA 0
#define LO_BETA 1
#define LO_F1 2
#define LO_F1B 3
#define LO_F2 4
#define LO_F2B 5
#define LO_N1B 6
#define LO_N1G 7
#define LO_N2B 8
#define LO_N2G 9
#define LO_WRPE 10
#define LO_WK 11
#define LO_WO 12
#define LO_WQ 13
#define LO_WV 14
#define IX_MLPB(i) (71 + (i))
#define IX_MLPBT(i) (76 + (i))
#define IX_MLPG(i) (80 + (i))
#define IX_MLPW(i) (84 + (i))

// =====================================================================
// pad x -> [NPAD,16], coords -> [NPAD,3] (zeros past N_RAW)
// =====================================================================
__global__ void pad_kernel(const float* __restrict__ x, const float* __restrict__ coords,
                           float* __restrict__ xpad, float* __restrict__ cpad) {
  int n = blockIdx.x * blockDim.x + threadIdx.x;
  if (n >= NPAD) return;
  bool live = n < N_RAW;
  for (int i = 0; i < IN_DIM; ++i) xpad[n * IN_DIM + i] = live ? x[n * IN_DIM + i] : 0.f;
  for (int i = 0; i < CD;     ++i) cpad[n * CD + i]     = live ? coords[n * CD + i] : 0.f;
}

// =====================================================================
// Generic WMMA GEMM: C[M,N] = op(A[M,K] @ B[K,N] + bias + resid)
// Requirements used for branch-free tiles: M % 64 == 0, K % 16 == 0, N % 16 == 0.
// block = 128 threads (4 waves); tile = 64(M) x 32(N); K stepped by 32.
// B tile stored transposed ([n][k]) so all fragment feeds are 16B LDS loads.
// out_f16: store _Float16 (for q/k/v); oscale folds 1/sqrt(HDIM) into wq.
// =====================================================================
__global__ __launch_bounds__(128)
void gemm_wmma(const float* __restrict__ A, int lda,
               const float* __restrict__ B, int ldb,
               const float* __restrict__ bias,
               const float* __restrict__ resid, int ldr,
               float* __restrict__ C, int ldc,
               int M, int N, int K, int do_relu, float oscale, int out_f16) {
  __shared__ alignas(16) _Float16 As[64 * 32];    // [m][k]
  __shared__ alignas(16) _Float16 BsT[32 * 32];   // [n][k]
  const int tid  = threadIdx.x;
  const int wave = tid >> 5;
  const int lane = tid & 31;
  const int bm0 = blockIdx.x * 64;
  const int bn0 = blockIdx.y * 32;

  v8f acc0 = {};
  v8f acc1 = {};

  for (int kc = 0; kc < K; kc += 32) {
    // ---- A tile: 64x32, float4 groups; gk<K is a whole-group predicate ----
    for (int g = tid; g < 512; g += 128) {
      int r = g >> 3, c4 = (g & 7) << 2;
      int gm = bm0 + r, gk = kc + c4;
      v4h hv = {};
      if (gk < K) {
        float4 f = *(const float4*)(A + (size_t)gm * lda + gk);
        hv[0] = (_Float16)f.x; hv[1] = (_Float16)f.y; hv[2] = (_Float16)f.z; hv[3] = (_Float16)f.w;
      }
      *(v4h*)(As + r * 32 + c4) = hv;
    }
    // ---- B tile: 32x32, loaded row-major, stored transposed ----
    for (int g = tid; g < 256; g += 128) {
      int r = g >> 3, c4 = (g & 7) << 2;          // r = k-local, c4 = n-local
      int gk = kc + r, gn = bn0 + c4;
      float4 f = make_float4(0.f, 0.f, 0.f, 0.f);
      if (gk < K && gn < N) f = *(const float4*)(B + (size_t)gk * ldb + gn);
      BsT[(c4 + 0) * 32 + r] = (_Float16)f.x;
      BsT[(c4 + 1) * 32 + r] = (_Float16)f.y;
      BsT[(c4 + 2) * 32 + r] = (_Float16)f.z;
      BsT[(c4 + 3) * 32 + r] = (_Float16)f.w;
    }
    __syncthreads();

    // A fragment (ISA 7.12.2): halfs 0..7 -> K = kb..kb+7, halfs 8..15 -> K = 16+kb..
    const int am = (wave << 4) + (lane & 15);
    const int kb = (lane >> 4) << 3;
    const _Float16* ap = As + am * 32 + kb;
    v16h af = cat16(*(const v8h*)ap, *(const v8h*)(ap + 16));

    const _Float16* bp0 = BsT + (lane & 15) * 32 + ((lane >> 4) << 3);
    v16h bf0 = cat16(*(const v8h*)bp0, *(const v8h*)(bp0 + 16));
    v16h bf1 = cat16(*(const v8h*)(bp0 + 512), *(const v8h*)(bp0 + 528));
    // same A operand back-to-back: RA reuse hint on the first WMMA
    acc0 = __builtin_amdgcn_wmma_f32_16x16x32_f16(false, af, false, bf0, (short)0, acc0, true,  false);
    acc1 = __builtin_amdgcn_wmma_f32_16x16x32_f16(false, af, false, bf1, (short)0, acc1, false, false);
    __syncthreads();
  }

  // epilogue: lane L reg r -> m = r + (L/16)*8, n = L%16  (M%64==0: no m guard)
  const int mb = bm0 + (wave << 4) + ((lane >> 4) << 3);
  #pragma unroll
  for (int t = 0; t < 2; ++t) {
    v8f acc = t ? acc1 : acc0;
    const int n = bn0 + (t << 4) + (lane & 15);
    if (n < N) {
      #pragma unroll
      for (int r = 0; r < 8; ++r) {
        int m = mb + r;
        float v = acc[r];
        if (bias)  v += bias[n];
        if (resid) v += resid[(size_t)m * ldr + n];
        if (do_relu) v = fmaxf(v, 0.f);
        v *= oscale;
        if (out_f16) ((_Float16*)C)[(size_t)m * ldc + n] = (_Float16)v;
        else         C[(size_t)m * ldc + n] = v;
      }
    }
  }
}

// =====================================================================
// wave-per-row LayerNorm (+ optional tanh); D must be a multiple of 32
// =====================================================================
__global__ __launch_bounds__(128)
void ln_kernel(const float* __restrict__ in, int ldin,
               const float* __restrict__ g, const float* __restrict__ b,
               float* __restrict__ out, int ldout,
               int M, int D, int do_tanh) {
  int row  = blockIdx.x * 4 + (threadIdx.x >> 5);
  int lane = threadIdx.x & 31;
  if (row >= M) return;
  const float* x = in + (size_t)row * ldin;
  float xr[8];
  const int nr = D >> 5;
  float sum = 0.f;
  for (int i = 0; i < nr; ++i) { xr[i] = x[lane + (i << 5)]; sum += xr[i]; }
  for (int off = 16; off > 0; off >>= 1) sum += __shfl_xor(sum, off, 32);
  float mean = sum / (float)D;
  float var = 0.f;
  for (int i = 0; i < nr; ++i) { float d = xr[i] - mean; var += d * d; }
  for (int off = 16; off > 0; off >>= 1) var += __shfl_xor(var, off, 32);
  var /= (float)D;
  float inv = rsqrtf(var + 1e-5f);
  float* y = out + (size_t)row * ldout;
  for (int i = 0; i < nr; ++i) {
    int c = lane + (i << 5);
    float v = (xr[i] - mean) * inv * g[c] + b[c];
    y[c] = do_tanh ? tanhf(v) : v;
  }
}

// =====================================================================
// w_hc[h,c] = mean_{d,kw} w_rpe[(h*32+d)*16 + c*8 + kw]^2   (16 outputs)
// =====================================================================
__global__ void whc_kernel(const float* __restrict__ w_rpe, float* __restrict__ whc) {
  int hc = blockIdx.x;              // 0..15
  int h = hc >> 1, c = hc & 1;
  __shared__ float red[256];
  int tid = threadIdx.x;            // 256 threads: d = tid>>3, kw = tid&7
  int d = tid >> 3, kw = tid & 7;
  float w = w_rpe[(h * HDIM + d) * ((CD - 1) * KW) + c * KW + kw];
  red[tid] = w * w;
  __syncthreads();
  for (int s = 128; s > 0; s >>= 1) {
    if (tid < s) red[tid] += red[tid + s];
    __syncthreads();
  }
  if (tid == 0) whc[hc] = red[0] / 256.f;
}

// =====================================================================
// sort keys from f16 q/k: key = bucket + frac(E2LSH hash) + pad-penalty
// =====================================================================
__global__ void keys_kernel(const _Float16* __restrict__ q, const _Float16* __restrict__ k,
                            const float* __restrict__ cpad,
                            const int* __restrict__ bin_e, const int* __restrict__ bin_p,
                            const float* __restrict__ alpha, const float* __restrict__ beta,
                            unsigned* __restrict__ skey, int* __restrict__ sidx) {
  int n = blockIdx.x * blockDim.x + threadIdx.x;
  if (n >= SORTN) return;
  if (n >= NPAD) {
    for (int s = 0; s < NSLICE; ++s) {
      skey[(size_t)s * SORTN + n] = 0x7FFFFFFFu;   // > +inf as uint: sorts last
      sidx[(size_t)s * SORTN + n] = n;
    }
    return;
  }
  float pad = (n >= N_RAW) ? 1e9f : 0.f;
  float c0 = cpad[n * CD + 0], c1 = cpad[n * CD + 1], c2 = cpad[n * CD + 2];
  for (int ah = 0; ah < NH * H; ++ah) {
    int h = ah & (H - 1);
    const float* al = alpha + ah * (HDIM + CD);
    float hq = beta[ah], hk = beta[ah];
    for (int e = 0; e < HDIM; ++e) {
      float w = al[e];
      hq += (float)q[(size_t)n * (H * HDIM) + h * HDIM + e] * w;
      hk += (float)k[(size_t)n * (H * HDIM) + h * HDIM + e] * w;
    }
    float wc = al[HDIM] * c0 + al[HDIM + 1] * c1 + al[HDIM + 2] * c2;
    hq += wc; hk += wc;
    float bucket = (float)(bin_e[(size_t)ah * NPAD + n] * 1024 + bin_p[(size_t)ah * NPAD + n]);
    float kq = bucket + (hq - floorf(hq)) + pad;
    float kk = bucket + (hk - floorf(hk)) + pad;
    skey[(size_t)ah * SORTN + n] = __float_as_uint(kq);           // all keys >= 0
    sidx[(size_t)ah * SORTN + n] = n;
    skey[(size_t)(NH * H + ah) * SORTN + n] = __float_as_uint(kk);
    sidx[(size_t)(NH * H + ah) * SORTN + n] = n;
  }
}

// =====================================================================
// one bitonic compare-exchange stage over all 48 slices (ascending overall)
// =====================================================================
__global__ void bitonic_step(unsigned* __restrict__ key, int* __restrict__ idx, int j, int kk) {
  int s = blockIdx.y;
  int i = blockIdx.x * blockDim.x + threadIdx.x;
  unsigned* Kp = key + (size_t)s * SORTN;
  int*      Ip = idx + (size_t)s * SORTN;
  int ixj = i ^ j;
  if (ixj > i) {
    bool up = ((i & kk) == 0);
    unsigned a = Kp[i], b = Kp[ixj];
    if ((a > b) == up) {
      Kp[i] = b; Kp[ixj] = a;
      int t = Ip[i]; Ip[i] = Ip[ixj]; Ip[ixj] = t;
    }
  }
}

// =====================================================================
// fused block attention: one block per (blk, h, a); 128 threads / 4 waves
// f16 q/k/v gathered via CDNA5 async-to-LDS (ASYNCcnt) when available
// =====================================================================
__global__ __launch_bounds__(128)
void attn_kernel(const _Float16* __restrict__ q, const _Float16* __restrict__ k,
                 const _Float16* __restrict__ v, const float* __restrict__ cpad,
                 const int* __restrict__ sidx, const float* __restrict__ whc,
                 float* __restrict__ o_accum) {
  const int blk = blockIdx.x, h = blockIdx.y, a = blockIdx.z;
  const int ah = a * H + h;
  const int* qi_ptr = sidx + (size_t)ah * SORTN + blk * BS;
  const int* ki_ptr = sidx + (size_t)(NH * H + ah) * SORTN + blk * BS;

  __shared__ alignas(16) _Float16 qf[BS * HDIM];     // [m][k]
  __shared__ alignas(16) _Float16 kf[BS * HDIM];     // [n][k] (B-major for S)
  __shared__ alignas(16) _Float16 vf[BS * HDIM];     // [row][k] gathered V
  __shared__ alignas(16) _Float16 vfT[HDIM * BS];    // [n][k] transposed V for O=P@V
  __shared__ float    S[BS * BS];
  __shared__ alignas(16) _Float16 P[BS * BS];
  __shared__ float cqs[BS * 2], cks[BS * 2];
  __shared__ int qidx[BS], kidxs[BS], kpad[BS];

  const int tid = threadIdx.x;
  const int wave = tid >> 5, lane = tid & 31;

  if (tid < BS) {
    int qi = qi_ptr[tid], ki = ki_ptr[tid];
    qidx[tid] = qi; kidxs[tid] = ki;
    kpad[tid] = (ki >= N_RAW);
    cqs[tid * 2 + 0] = cpad[qi * CD + 0]; cqs[tid * 2 + 1] = cpad[qi * CD + 1];
    cks[tid * 2 + 0] = cpad[ki * CD + 0]; cks[tid * 2 + 1] = cpad[ki * CD + 1];
  }
  __syncthreads();

  // ---- gather: 64 rows x 32 halfs per matrix, 16B per lane-op ----
#if HAVE_ASYNC_LDS
  for (int g = tid; g < 256; g += 128) {
    int r = g >> 2, c8 = (g & 3) << 3;
    int qi = qidx[r], ki = kidxs[r];
    size_t coff = (size_t)h * HDIM + c8;
    ASYNC_B128(q + (size_t)qi * (H * HDIM) + coff, qf + r * HDIM + c8);
    ASYNC_B128(k + (size_t)ki * (H * HDIM) + coff, kf + r * HDIM + c8);
    ASYNC_B128(v + (size_t)ki * (H * HDIM) + coff, vf + r * HDIM + c8);
  }
#  if __has_builtin(__builtin_amdgcn_s_wait_asynccnt)
  __builtin_amdgcn_s_wait_asynccnt(0);
#  else
  asm volatile("s_wait_asynccnt 0x0" ::: "memory");
#  endif
#else
  for (int g = tid; g < 256; g += 128) {
    int r = g >> 2, c8 = (g & 3) << 3;
    int qi = qidx[r], ki = kidxs[r];
    size_t coff = (size_t)h * HDIM + c8;
    *(v8h*)(qf + r * HDIM + c8) = *(const v8h*)(q + (size_t)qi * (H * HDIM) + coff);
    *(v8h*)(kf + r * HDIM + c8) = *(const v8h*)(k + (size_t)ki * (H * HDIM) + coff);
    *(v8h*)(vf + r * HDIM + c8) = *(const v8h*)(v + (size_t)ki * (H * HDIM) + coff);
  }
#endif
  __syncthreads();
  // LDS transpose of V (raw byte moves; no conversion needed)
  for (int g = tid; g < BS * HDIM; g += 128) {
    int r = g >> 5, c = g & 31;
    vfT[c * BS + r] = vf[r * HDIM + c];
  }
  __syncthreads();

  // ---- S = q @ k^T (per-wave 16-row stripe, 4 column tiles) ----
  const int am = (wave << 4) + (lane & 15);
  const int kb = (lane >> 4) << 3;
  const _Float16* ap = qf + am * HDIM + kb;
  v16h af = cat16(*(const v8h*)ap, *(const v8h*)(ap + 16));

  const float w0 = whc[h * 2 + 0], w1 = whc[h * 2 + 1];
  const int mbl = (wave << 4) + ((lane >> 4) << 3);

#define S_TILE(J, RA)                                                                         \
  {                                                                                           \
    const int n = ((J) << 4) + (lane & 15);                                                   \
    const _Float16* bp = kf + n * HDIM + ((lane >> 4) << 3);                                  \
    v16h bf = cat16(*(const v8h*)bp, *(const v8h*)(bp + 16));                                 \
    v8f acc = {};                                                                             \
    acc = __builtin_amdgcn_wmma_f32_16x16x32_f16(false, af, false, bf, (short)0, acc, RA, false); \
    _Pragma("unroll")                                                                         \
    for (int r = 0; r < 8; ++r) {                                                             \
      int m = mbl + r;                                                                        \
      float dx = cqs[m * 2 + 0] - cks[n * 2 + 0];                                             \
      float dy = cqs[m * 2 + 1] - cks[n * 2 + 1];                                             \
      float s = acc[r] - (w0 * dx * dx + w1 * dy * dy);                                       \
      if (kpad[n]) s = -1e9f;                                                                 \
      S[m * BS + n] = s;                                                                      \
    }                                                                                         \
  }
  S_TILE(0, true)
  S_TILE(1, true)
  S_TILE(2, true)
  S_TILE(3, false)
#undef S_TILE
  __syncthreads();

  // ---- softmax rows (one thread per row) ----
  if (tid < BS) {
    float mx = -1e30f;
    for (int n = 0; n < BS; ++n) mx = fmaxf(mx, S[tid * BS + n]);
    float sum = 0.f;
    for (int n = 0; n < BS; ++n) { float e = expf(S[tid * BS + n] - mx); S[tid * BS + n] = e; sum += e; }
    float inv = 1.f / sum;
    for (int n = 0; n < BS; ++n) P[tid * BS + n] = (_Float16)(S[tid * BS + n] * inv);
  }
  __syncthreads();

  // ---- O = P @ V : 64x64 @ 64x32, K split into two 32-chunks ----
  #pragma unroll
  for (int t = 0; t < 2; ++t) {
    v8f acc = {};
    #pragma unroll
    for (int kc = 0; kc < 2; ++kc) {
      const _Float16* pp = P + am * BS + (kc << 5) + kb;
      v16h pa = cat16(*(const v8h*)pp, *(const v8h*)(pp + 16));
      const int n = (t << 4) + (lane & 15);
      const _Float16* bp = vfT + n * BS + (kc << 5) + ((lane >> 4) << 3);
      v16h bf = cat16(*(const v8h*)bp, *(const v8h*)(bp + 16));
      acc = __builtin_amdgcn_wmma_f32_16x16x32_f16(false, pa, false, bf, (short)0, acc, false, false);
    }
    const int n = (t << 4) + (lane & 15);
    #pragma unroll
    for (int r = 0; r < 8; ++r) {
      int m = mbl + r;
      int orig = qidx[m];
      unsafeAtomicAdd(&o_accum[(size_t)orig * (H * HDIM) + h * HDIM + n], acc[r] * (1.f / (float)NH));
    }
  }
}

// =====================================================================
// out[:N_RAW] = z + m
// =====================================================================
__global__ void final_add(const float* __restrict__ z, const float* __restrict__ m,
                          float* __restrict__ out) {
  int i = blockIdx.x * blockDim.x + threadIdx.x;
  if (i >= N_RAW * HD2) return;
  out[i] = z[i] + m[i];
}

// =====================================================================
// host driver
// =====================================================================
static inline void launch_gemm(const float* A, int lda, const float* B, int ldb,
                               const float* bias, const float* resid, int ldr,
                               float* C, int ldc, int M, int N, int K, int relu,
                               float oscale, int out_f16, hipStream_t s) {
  dim3 g((M + 63) / 64, (N + 31) / 32);
  gemm_wmma<<<g, 128, 0, s>>>(A, lda, B, ldb, bias, resid, ldr, C, ldc, M, N, K, relu, oscale, out_f16);
}

extern "C" void kernel_launch(void* const* d_in, const int* in_sizes, int n_in,
                              void* d_out, int out_size, void* d_ws, size_t ws_size,
                              hipStream_t stream) {
  (void)in_sizes; (void)n_in; (void)out_size; (void)ws_size;
  auto F = [&](int i) { return (const float*)d_in[i]; };
  auto I = [&](int i) { return (const int*)d_in[i]; };

  const float* x      = F(0);
  const float* coords = F(1);
  const int*   bin_e  = I(4);
  const int*   bin_p  = I(5);

  float* ws = (float*)d_ws;
  size_t o = 0;
  float* xpad  = ws + o; o += (size_t)NPAD * IN_DIM;
  float* cpad  = ws + o; o += (size_t)NPAD * CD;
  float* halls = ws + o; o += (size_t)NPAD * (NL + 1) * HDIM;   // ld = 160
  float* xn    = ws + o; o += (size_t)NPAD * HDIM;
  float* hmid  = ws + o; o += (size_t)NPAD * HDIM;
  float* tbuf  = ws + o; o += (size_t)NPAD * HDIM;
  float* qb    = ws + o; o += (size_t)NPAD * (H * HDIM);   // also f16 q & MLP ping
  float* kb    = ws + o; o += (size_t)NPAD * (H * HDIM);   // also f16 k & MLP pong
  float* vb    = ws + o; o += (size_t)NPAD * (H * HDIM);   // also f16 v
  float* oacc  = ws + o; o += (size_t)NPAD * (H * HDIM);
  unsigned* skey  = (unsigned*)(ws + o); o += (size_t)NSLICE * SORTN;
  int*      sidxb = (int*)(ws + o);      o += (size_t)NSLICE * SORTN;
  float* whcbuf = ws + o; o += 16;
  float* zbuf   = ws + o; o += (size_t)NPAD * HD2;

  _Float16* qh = (_Float16*)qb;
  _Float16* kh = (_Float16*)kb;
  _Float16* vh = (_Float16*)vb;

  const int LDH = (NL + 1) * HDIM;   // 160
  const int LNG = (NPAD + 3) / 4;    // wave-per-row LN grid
  const float QSCALE = 0.1767766952966369f;  // 1/sqrt(32)

  // ---- pad + embedding ----
  pad_kernel<<<(NPAD + 255) / 256, 256, 0, stream>>>(x, coords, xpad, cpad);
  launch_gemm(xpad, IN_DIM, F(IX_E1), HDIM, F(IX_E1B), nullptr, 0,
              xn, HDIM, NPAD, HDIM, IN_DIM, 1, 1.f, 0, stream);
  launch_gemm(xn, HDIM, F(IX_E2), HDIM, F(IX_E2B), nullptr, 0,
              halls, LDH, NPAD, HDIM, HDIM, 0, 1.f, 0, stream);

  // ---- transformer layers ----
  for (int L = 0; L < NL; ++L) {
    const int b = IX_LAYER(L);
    const float* hprev = halls + (size_t)L * HDIM;   // ld = LDH
    float* hnext = halls + (size_t)(L + 1) * HDIM;

    whc_kernel<<<16, 256, 0, stream>>>(F(b + LO_WRPE), whcbuf);
    ln_kernel<<<LNG, 128, 0, stream>>>(hprev, LDH, F(b + LO_N1G), F(b + LO_N1B),
                                       xn, HDIM, NPAD, HDIM, 0);
    // q/k/v projections -> f16; fold 1/sqrt(HDIM) into q
    launch_gemm(xn, HDIM, F(b + LO_WQ), H * HDIM, nullptr, nullptr, 0, (float*)qh, H * HDIM,
                NPAD, H * HDIM, HDIM, 0, QSCALE, 1, stream);
    launch_gemm(xn, HDIM, F(b + LO_WK), H * HDIM, nullptr, nullptr, 0, (float*)kh, H * HDIM,
                NPAD, H * HDIM, HDIM, 0, 1.f, 1, stream);
    launch_gemm(xn, HDIM, F(b + LO_WV), H * HDIM, nullptr, nullptr, 0, (float*)vh, H * HDIM,
                NPAD, H * HDIM, HDIM, 0, 1.f, 1, stream);

    keys_kernel<<<SORTN / 256, 256, 0, stream>>>(qh, kh, cpad, bin_e, bin_p,
                                                 F(b + LO_ALPHA), F(b + LO_BETA), skey, sidxb);
    for (int kk = 2; kk <= SORTN; kk <<= 1)
      for (int j = kk >> 1; j > 0; j >>= 1)
        bitonic_step<<<dim3(SORTN / 256, NSLICE), 256, 0, stream>>>(skey, sidxb, j, kk);

    (void)hipMemsetAsync(oacc, 0, (size_t)NPAD * (H * HDIM) * sizeof(float), stream);
    attn_kernel<<<dim3(NBLK, H, NH), 128, 0, stream>>>(qh, kh, vh, cpad, sidxb, whcbuf, oacc);

    launch_gemm(oacc, H * HDIM, F(b + LO_WO), HDIM, nullptr, hprev, LDH,
                hmid, HDIM, NPAD, HDIM, H * HDIM, 0, 1.f, 0, stream);
    ln_kernel<<<LNG, 128, 0, stream>>>(hmid, HDIM, F(b + LO_N2G), F(b + LO_N2B),
                                       xn, HDIM, NPAD, HDIM, 0);
    launch_gemm(xn, HDIM, F(b + LO_F1), HDIM, F(b + LO_F1B), nullptr, 0,
                tbuf, HDIM, NPAD, HDIM, HDIM, 1, 1.f, 0, stream);
    launch_gemm(tbuf, HDIM, F(b + LO_F2), HDIM, F(b + LO_F2B), hmid, HDIM,
                hnext, LDH, NPAD, HDIM, HDIM, 0, 1.f, 0, stream);
  }

  // ---- concat projection: z = halls[NPAD,160] @ W[160,16] ----
  launch_gemm(halls, LDH, F(IX_W), HD2, nullptr, nullptr, 0, zbuf, HD2,
              NPAD, HD2, LDH, 0, 1.f, 0, stream);

  // ---- MLP head (reuse qb/kb/oacc as MH-wide ping-pong) ----
  const float* min_ptr = zbuf; int ldin = HD2; int Kd = HD2;
  float* lnout[2] = { qb, kb };
  for (int i = 0; i < 4; ++i) {
    launch_gemm(min_ptr, ldin, F(IX_MLPW(i)), MH, F(IX_MLPB(i)), nullptr, 0,
                oacc, MH, NPAD, MH, Kd, 0, 1.f, 0, stream);
    ln_kernel<<<LNG, 128, 0, stream>>>(oacc, MH, F(IX_MLPG(i)), F(IX_MLPBT(i)),
                                       lnout[i & 1], MH, NPAD, MH, 1);
    min_ptr = lnout[i & 1]; ldin = MH; Kd = MH;
  }
  launch_gemm(min_ptr, MH, F(IX_MLPW(4)), HD2, F(IX_MLPB(4)), nullptr, 0,
              tbuf, HD2, NPAD, HD2, MH, 0, 1.f, 0, stream);

  final_add<<<(N_RAW * HD2 + 255) / 256, 256, 0, stream>>>(zbuf, tbuf, (float*)d_out);
}
